// StereoCorrespondencesLoss_71691594105404
// MI455X (gfx1250) — matrix operand order repaired and checked by reading, not compiled, not run on hardware
//
#include <hip/hip_runtime.h>

typedef __attribute__((ext_vector_type(16))) _Float16 v16h;
typedef __attribute__((ext_vector_type(8)))  _Float16 v8h;
typedef __attribute__((ext_vector_type(8)))  float    v8f;

#define Cc   128
#define Nn   8192
#define Hh   256
#define Ww   256
#define Bb   2
// 1/(TEMPERATURE * ln2) = (1/0.07) * log2(e): baked into d1 so WMMA outputs
// are directly exp2 logits -> zero multiplies in the hot loop.
#define A_SCALE 20.609929155556625f
#define EPSL 1e-8f

// ---------------------------------------------------------------------------
// 1) Gather descriptors at pixel coords, f32 -> f16, store [B][N][C] row-major.
//    d1 is pre-scaled by 1/(T*ln2); d2 stays raw.
// ---------------------------------------------------------------------------
__global__ void gather_kernel(const float* __restrict__ desc1,
                              const float* __restrict__ desc2,
                              const int*   __restrict__ pc1,
                              const int*   __restrict__ pc2,
                              _Float16* __restrict__ d1h,
                              _Float16* __restrict__ d2h) {
    int bn = blockIdx.x;             // b*N + n
    int c  = threadIdx.x;            // 0..127
    int b  = bn >> 13;               // / 8192
    int x1 = pc1[bn * 2 + 0], y1 = pc1[bn * 2 + 1];
    int x2 = pc2[bn * 2 + 0], y2 = pc2[bn * 2 + 1];
    size_t plane = (size_t)Hh * Ww;
    size_t base  = ((size_t)b * Cc + c) * plane;
    float v1 = desc1[base + (size_t)y1 * Ww + x1];
    float v2 = desc2[base + (size_t)y2 * Ww + x2];
    d1h[(size_t)bn * Cc + c] = (_Float16)(v1 * A_SCALE);
    d2h[(size_t)bn * Cc + c] = (_Float16)v2;
}

// ---------------------------------------------------------------------------
// 2) positives[i] = exp2(dot(d1_scaled[i], d2[i]))   (one wave32 per point)
// ---------------------------------------------------------------------------
__global__ void pos_kernel(const _Float16* __restrict__ d1h,
                           const _Float16* __restrict__ d2h,
                           float* __restrict__ pos) {
    int gwave = (blockIdx.x * blockDim.x + threadIdx.x) >> 5;
    int lane  = threadIdx.x & 31;
    if (gwave >= Bb * Nn) return;
    const _Float16* a = d1h + (size_t)gwave * Cc;
    const _Float16* b = d2h + (size_t)gwave * Cc;
    float s = 0.f;
    for (int k = lane; k < Cc; k += 32)
        s += (float)a[k] * (float)b[k];
    for (int m = 16; m >= 1; m >>= 1)
        s += __shfl_xor(s, m, 32);
    if (lane == 0) pos[gwave] = __builtin_amdgcn_exp2f(s);
}

// ---------------------------------------------------------------------------
// 3) Fused WMMA exp2-row-sum: out[i] = sum_j exp2(dot(A_i, B_j)).
//    pass 0: A=d1s,B=d2 -> rowsum ; pass 1: A=d2,B=d1s -> colsum (transpose).
//    4 waves/block, 32 rows per wave (two A tiles per B tile -> 2x reuse),
//    loop over all 512 column tiles.
// ---------------------------------------------------------------------------
__global__ __launch_bounds__(128) void expsum_kernel(
    const _Float16* __restrict__ d1h, const _Float16* __restrict__ d2h,
    float* __restrict__ rowsum, float* __restrict__ colsum) {
    int b    = blockIdx.y;
    int pass = blockIdx.z;
    const _Float16* A  = (pass == 0 ? d1h : d2h) + (size_t)b * Nn * Cc;
    const _Float16* Bm = (pass == 0 ? d2h : d1h) + (size_t)b * Nn * Cc;
    float* out         = (pass == 0 ? rowsum : colsum) + (size_t)b * Nn;

    int lane = threadIdx.x & 31;
    int wave = threadIdx.x >> 5;
    int rowBase = (blockIdx.x * 4 + wave) * 32;   // this wave owns 32 rows
    int hi  = lane >> 4;     // 0: lanes 0-15, 1: lanes 16-31
    int lid = lane & 15;

    // A operands (16x32 f16 per K-chunk), loop-invariant -> resident in VGPRs.
    // lanes 0-15 : h0..7 = K k0..k0+7   , h8..15 = K k0+16..k0+23
    // lanes 16-31: h0..7 = K k0+8..k0+15, h8..15 = K k0+24..k0+31
    v16h a0[4], a1[4];
    {
        const _Float16* arow0 = A + (size_t)(rowBase + lid) * Cc;
        const _Float16* arow1 = arow0 + (size_t)16 * Cc;
        #pragma unroll
        for (int kk = 0; kk < 4; ++kk) {
            int k0 = kk * 32 + hi * 8;
            v8h lo0 = *(const v8h*)(arow0 + k0);
            v8h up0 = *(const v8h*)(arow0 + k0 + 16);
            v8h lo1 = *(const v8h*)(arow1 + k0);
            v8h up1 = *(const v8h*)(arow1 + k0 + 16);
            a0[kk] = __builtin_shufflevector(lo0, up0,
                       0,1,2,3,4,5,6,7,8,9,10,11,12,13,14,15);
            a1[kk] = __builtin_shufflevector(lo1, up1,
                       0,1,2,3,4,5,6,7,8,9,10,11,12,13,14,15);
        }
    }

    float acc0[8] = {0.f,0.f,0.f,0.f,0.f,0.f,0.f,0.f};
    float acc1[8] = {0.f,0.f,0.f,0.f,0.f,0.f,0.f,0.f};

    for (int tile = 0; tile < Nn; tile += 16) {
        // B operand (32x16): lane holds K 8h..8h+7 and 16+8h..16+8h+7 of one d2-row (column).
        const _Float16* brow = Bm + (size_t)(tile + lid) * Cc + hi * 8;
        v16h bv[4];
        #pragma unroll
        for (int kk = 0; kk < 4; ++kk)
            bv[kk] = __builtin_shufflevector(*(const v8h*)(brow + kk * 32), *(const v8h*)(brow + kk * 32 + 16), 0,1,2,3,4,5,6,7,8,9,10,11,12,13,14,15);

        v8f c0 = {}, c1 = {};
        #pragma unroll
        for (int kk = 0; kk < 4; ++kk)
            c0 = __builtin_amdgcn_wmma_f32_16x16x32_f16(
                     false, a0[kk], false, bv[kk], (short)0, c0, false, false);
        #pragma unroll
        for (int kk = 0; kk < 4; ++kk)
            c1 = __builtin_amdgcn_wmma_f32_16x16x32_f16(
                     false, a1[kk], false, bv[kk], (short)0, c1, false, false);

        // C/D layout: VGPR v -> row v (lanes 0-15) / row v+8 (lanes 16-31).
        // Logits already scaled by 1/(T*ln2): single v_exp per element.
        #pragma unroll
        for (int v = 0; v < 8; ++v) {
            acc0[v] += __builtin_amdgcn_exp2f(c0[v]);
            acc1[v] += __builtin_amdgcn_exp2f(c1[v]);
        }
    }

    // Reduce each 16-lane half (each half owns a distinct set of 8 rows).
    #pragma unroll
    for (int v = 0; v < 8; ++v) {
        float s0 = acc0[v], s1 = acc1[v];
        for (int m = 8; m >= 1; m >>= 1) {
            s0 += __shfl_xor(s0, m, 16);
            s1 += __shfl_xor(s1, m, 16);
        }
        acc0[v] = s0; acc1[v] = s1;
    }
    if (lid == 0) {
        #pragma unroll
        for (int v = 0; v < 8; ++v) {
            out[rowBase + v + hi * 8]      = acc0[v];
            out[rowBase + 16 + v + hi * 8] = acc1[v];
        }
    }
}

// ---------------------------------------------------------------------------
// 4) loss = sum_i -log(max(pos^2/(rowsum*colsum), EPS)) / (1+EPS)
// ---------------------------------------------------------------------------
__global__ void loss_kernel(const float* __restrict__ pos,
                            const float* __restrict__ rowsum,
                            const float* __restrict__ colsum,
                            float* __restrict__ out) {
    __shared__ float red[256];
    float s = 0.f;
    for (int i = threadIdx.x; i < Bb * Nn; i += 256) {
        float p = pos[i];
        float r = (p * p) / (rowsum[i] * colsum[i]);
        s -= __logf(fmaxf(r, EPSL));
    }
    red[threadIdx.x] = s;
    __syncthreads();
    for (int off = 128; off > 0; off >>= 1) {
        if (threadIdx.x < off) red[threadIdx.x] += red[threadIdx.x + off];
        __syncthreads();
    }
    if (threadIdx.x == 0)
        out[0] = red[0] * (1.0f / (1.0f + EPSL));   // * LOSS_WEIGHT (1.0)
}

// ---------------------------------------------------------------------------
extern "C" void kernel_launch(void* const* d_in, const int* in_sizes, int n_in,
                              void* d_out, int out_size, void* d_ws, size_t ws_size,
                              hipStream_t stream) {
    (void)in_sizes; (void)n_in; (void)out_size; (void)ws_size;
    const float* desc1 = (const float*)d_in[0];
    const float* desc2 = (const float*)d_in[1];
    const int*   pc1   = (const int*)d_in[2];
    const int*   pc2   = (const int*)d_in[3];

    char* ws = (char*)d_ws;
    size_t descBytes = (size_t)Bb * Nn * Cc * 2;      // 4 MB each
    _Float16* d1h   = (_Float16*)(ws);
    _Float16* d2h   = (_Float16*)(ws + descBytes);
    float*    pos    = (float*)(ws + 2 * descBytes);
    float*    rowsum = pos + Bb * Nn;
    float*    colsum = rowsum + Bb * Nn;

    gather_kernel<<<dim3(Bb * Nn), dim3(Cc), 0, stream>>>(desc1, desc2, pc1, pc2, d1h, d2h);
    pos_kernel  <<<dim3(Bb * Nn / 8), dim3(256), 0, stream>>>(d1h, d2h, pos);
    expsum_kernel<<<dim3(Nn / 128, Bb, 2), dim3(128), 0, stream>>>(d1h, d2h, rowsum, colsum);
    loss_kernel <<<dim3(1), dim3(256), 0, stream>>>(pos, rowsum, colsum, (float*)d_out);
}
